// MHA_86818468922108
// MI455X (gfx1250) — matrix operand order repaired and checked by reading, not compiled, not run on hardware
//
#include <hip/hip_runtime.h>

// Problem constants (match reference)
#define BB   4
#define LL   2048
#define DD   1024
#define HH   16
#define DKK  64
#define MM   (BB * LL)   // 8192 rows total

typedef __attribute__((ext_vector_type(16))) _Float16 v16h;
typedef __attribute__((ext_vector_type(8)))  _Float16 v8h;
typedef __attribute__((ext_vector_type(8)))  float    v8f;

// ---------------------------------------------------------------------------
// WMMA wrapper: D = A(16x32 f16) * B(32x16 f16) + C(16x16 f32)
// ---------------------------------------------------------------------------
__device__ __forceinline__ v8f wmma_f16(v16h a, v16h b, v8f c) {
  return __builtin_amdgcn_wmma_f32_16x16x32_f16(
      /*neg_a=*/false, a, /*neg_b=*/false, b,
      /*c_mod=*/(short)0, c, /*reuse_a=*/false, /*reuse_b=*/false);
}

// ---------------------------------------------------------------------------
// Fragment loaders.  16-bit A layout used here, contiguous k (16x32, MxK):
//   lanes 0-15 : M=lane,    K = 0..15
//   lanes 16-31: M=lane-16, K = 16..31
// B layout (32x16, KxN): lane<16 -> N=lane, K=0..15 ; lane>=16 -> N=lane-16, K=16..31
// ---------------------------------------------------------------------------
__device__ __forceinline__ v16h load_a_f32(const float* __restrict__ base, int ld, int lane) {
  const int row  = lane & 15;
  const int kofs = (lane >> 4) << 4;          // 0 or 16
  const float* p = base + (size_t)row * ld + kofs;
  v8f lo = *(const v8f*)p;                    // K = kofs .. kofs+7
  v8f hi = *(const v8f*)(p + 8);              // K = kofs+8 .. kofs+15
  v16h r;
#pragma unroll
  for (int i = 0; i < 8; ++i) { r[i] = (_Float16)lo[i]; r[i + 8] = (_Float16)hi[i]; }
  return r;
}

__device__ __forceinline__ v16h load_a_f16(const _Float16* base, int ld, int lane) {
  const int row  = lane & 15;
  const int kofs = (lane >> 4) << 4;
  const _Float16* p = base + (size_t)row * ld + kofs;
  v8h lo = *(const v8h*)p;
  v8h hi = *(const v8h*)(p + 8);
  v16h r;
#pragma unroll
  for (int i = 0; i < 8; ++i) { r[i] = lo[i]; r[i + 8] = hi[i]; }
  return r;
}

__device__ __forceinline__ v16h load_b_f16(const _Float16* base, int ld, int lane) {
  const int col  = lane & 15;
  const int kofs = (lane >> 4) << 4;
  return *(const v16h*)(base + (size_t)col * ld + kofs);   // 32 contiguous bytes
}

// ---------------------------------------------------------------------------
// fp32 -> f16 conversion pre-pass (weights only; 8 elements/thread)
// ---------------------------------------------------------------------------
__global__ __launch_bounds__(256) void cvt_f32_f16_kernel(
    const float* __restrict__ src, _Float16* __restrict__ dst, int n) {
  const int i = (blockIdx.x * 256 + threadIdx.x) * 8;
  if (i + 8 <= n) {
    v8f x = *(const v8f*)(src + i);
    v8h y;
#pragma unroll
    for (int j = 0; j < 8; ++j) y[j] = (_Float16)x[j];
    *(v8h*)(dst + i) = y;
  }
}

// ---------------------------------------------------------------------------
// GEMM: Y[M,N] = A[M,K](f32) @ W[N,K](f16)^T + bias[N]
// MODE 0: f16 row-major out     (Q, K projections)
// MODE 1: f16 transposed out -> [B, D, L]   (V projection)
// MODE 2: f32 row-major out     (output projection -> d_out)
// Block = 256 threads (8 waves); wave tile = 32(M) x 128(N); K-step = 32.
// Per K-chunk: 2 A-frags x 8 B-frags -> 16 WMMAs (every B-frag reused twice).
// ---------------------------------------------------------------------------
template <int MODE>
__global__ __launch_bounds__(256) void gemm_wmma_kernel(
    const float* __restrict__ A, const _Float16* __restrict__ Wh,
    const float* __restrict__ bias, _Float16* __restrict__ outH,
    float* __restrict__ outF, int Mdim, int Ndim, int Kdim) {
  const int lane = threadIdx.x & 31;
  const int wave = threadIdx.x >> 5;
  const int m0 = blockIdx.x * 256 + wave * 32;
  const int n0 = blockIdx.y * 128;

  v8f acc[2][8] = {};

  for (int k0 = 0; k0 < Kdim; k0 += 32) {
    // prefetch next K-chunk (speculative; OOB translation failures are dropped)
    __builtin_prefetch(A + (size_t)(m0 + lane) * Kdim + k0 + 32, 0, 3);
    __builtin_prefetch(Wh + (size_t)(n0 + lane * 4) * Kdim + k0 + 32, 0, 3);

    const v16h a0 = load_a_f32(A + (size_t)m0 * Kdim + k0, Kdim, lane);
    const v16h a1 = load_a_f32(A + (size_t)(m0 + 16) * Kdim + k0, Kdim, lane);
#pragma unroll
    for (int t = 0; t < 8; ++t) {
      const v16h b = load_b_f16(Wh + (size_t)(n0 + t * 16) * Kdim + k0, Kdim, lane);
      acc[0][t] = wmma_f16(a0, b, acc[0][t]);
      acc[1][t] = wmma_f16(a1, b, acc[1][t]);
    }
  }

  const int col   = lane & 15;
  const int rowhi = (lane >> 4) * 8;
#pragma unroll
  for (int t = 0; t < 8; ++t) {
    const int n  = n0 + t * 16 + col;
    const float bb = bias[n];
#pragma unroll
    for (int rt = 0; rt < 2; ++rt) {
#pragma unroll
      for (int r = 0; r < 8; ++r) {
        const float val = acc[rt][t][r] + bb;
        const int m = m0 + rt * 16 + rowhi + r;
        if (MODE == 0) {
          outH[(size_t)m * Ndim + n] = (_Float16)val;
        } else if (MODE == 1) {
          const int bt = m >> 11;                 // m / LL (LL = 2048)
          const int lt = m & (LL - 1);
          outH[((size_t)bt * Ndim + n) * LL + lt] = (_Float16)val;
        } else {
          outF[(size_t)m * Ndim + n] = val;
        }
      }
    }
  }
}

// ---------------------------------------------------------------------------
// Causal flash attention. One wave = 32 query rows (two 16-row tiles) of one
// (b,h); both row tiles share every K/V fragment.
// Qh, Kh: f16 [B*L, D] (head h at column h*64). Vt: f16 [B, D, L] transposed.
// O: f32 [B*L, D].
// Row sums of P come from a WMMA against a ones fragment (lands in C-layout).
// ---------------------------------------------------------------------------
__global__ __launch_bounds__(256) void flash_attn_kernel(
    const _Float16* __restrict__ Qh, const _Float16* __restrict__ Kh,
    const _Float16* __restrict__ Vt, float* __restrict__ O) {
  __shared__ __align__(32) _Float16 Plds[8][32][40];  // per-wave 32x32 P (+pad)

  const int lane = threadIdx.x & 31;
  const int wave = threadIdx.x >> 5;
  const int bh = blockIdx.y;
  const int b = bh >> 4;         // / HH
  const int h = bh & 15;
  const int q0 = (blockIdx.x * 8 + wave) * 32;

  const _Float16* Qp = Qh + ((size_t)(b * LL + q0)) * DD + h * DKK;
  const _Float16* Kp = Kh + ((size_t)b * LL) * DD + h * DKK;
  const _Float16* Vp = Vt + ((size_t)b * DD + h * DKK) * LL;

  // Q fragments: [row-tile][feature chunk], resident for the whole key loop
  v16h aq[2][2];
#pragma unroll
  for (int rt = 0; rt < 2; ++rt)
#pragma unroll
    for (int c = 0; c < 2; ++c)
      aq[rt][c] = load_a_f16(Qp + (size_t)(rt * 16) * DD + c * 32, DD, lane);

  v16h ones;
#pragma unroll
  for (int i = 0; i < 16; ++i) ones[i] = (_Float16)1.0f;

  v8f m_i[2], lacc[2] = {}, o[2][4] = {};
#pragma unroll
  for (int rt = 0; rt < 2; ++rt)
#pragma unroll
    for (int r = 0; r < 8; ++r) m_i[rt][r] = -1e30f;

  const int col   = lane & 15;
  const int rowhi = (lane >> 4) * 8;
  _Float16* pl = &Plds[wave][0][0];

  for (int kb = 0; kb < q0 + 32; kb += 32) {
    // ---- S(32x32) = Q(32x64) . K^T : 8 WMMAs on 4 K-fragments ----
    v8f s[2][2] = {};
#pragma unroll
    for (int t = 0; t < 2; ++t) {
#pragma unroll
      for (int c = 0; c < 2; ++c) {
        const v16h bk = load_b_f16(Kp + (size_t)(kb + 16 * t) * DD + c * 32, DD, lane);
        s[0][t] = wmma_f16(aq[0][c], bk, s[0][t]);
        s[1][t] = wmma_f16(aq[1][c], bk, s[1][t]);
      }
    }
#pragma unroll
    for (int rt = 0; rt < 2; ++rt) {
      s[rt][0] *= 0.125f;   // 1/sqrt(DK)
      s[rt][1] *= 0.125f;
    }

    // ---- causal mask (only the final key block needs it) ----
    if (kb + 31 > q0) {
#pragma unroll
      for (int rt = 0; rt < 2; ++rt)
#pragma unroll
        for (int r = 0; r < 8; ++r) {
          const int qa = q0 + rt * 16 + rowhi + r;
          s[rt][0][r] = (kb + col > qa)      ? -1e9f : s[rt][0][r];
          s[rt][1][r] = (kb + 16 + col > qa) ? -1e9f : s[rt][1][r];
        }
    }

    // ---- online softmax (row max across 16-lane half-waves) ----
    v8f alpha[2];
#pragma unroll
    for (int rt = 0; rt < 2; ++rt) {
      v8f mx;
#pragma unroll
      for (int r = 0; r < 8; ++r) mx[r] = fmaxf(s[rt][0][r], s[rt][1][r]);
#pragma unroll
      for (int sh = 1; sh < 16; sh <<= 1) {
#pragma unroll
        for (int r = 0; r < 8; ++r) mx[r] = fmaxf(mx[r], __shfl_xor(mx[r], sh, 32));
      }
#pragma unroll
      for (int r = 0; r < 8; ++r) {
        const float mnew = fmaxf(m_i[rt][r], mx[r]);
        alpha[rt][r] = __expf(m_i[rt][r] - mnew);
        m_i[rt][r]   = mnew;
      }
#pragma unroll
      for (int r = 0; r < 8; ++r) {
        s[rt][0][r] = __expf(s[rt][0][r] - m_i[rt][r]);
        s[rt][1][r] = __expf(s[rt][1][r] - m_i[rt][r]);
      }
    }

    // ---- stage both P tiles (C-layout -> A-layout) through per-wave LDS ----
#pragma unroll
    for (int rt = 0; rt < 2; ++rt)
#pragma unroll
      for (int r = 0; r < 8; ++r) {
        pl[(rt * 16 + rowhi + r) * 40 + col]      = (_Float16)s[rt][0][r];
        pl[(rt * 16 + rowhi + r) * 40 + 16 + col] = (_Float16)s[rt][1][r];
      }
    asm volatile("s_wait_dscnt 0" ::: "memory");  // cross-lane LDS RAW inside wave
    const v16h pa0 = load_a_f16(pl, 40, lane);
    const v16h pa1 = load_a_f16(pl + 16 * 40, 40, lane);

    // ---- rescale state, then 10 WMMAs: l += P.1 ; O += P.V ----
#pragma unroll
    for (int r = 0; r < 8; ++r) { lacc[0][r] *= alpha[0][r]; lacc[1][r] *= alpha[1][r]; }
    lacc[0] = wmma_f16(pa0, ones, lacc[0]);
    lacc[1] = wmma_f16(pa1, ones, lacc[1]);
#pragma unroll
    for (int f = 0; f < 4; ++f) {
      const v16h bv = load_b_f16(Vp + (size_t)(f * 16) * LL + kb, LL, lane);
#pragma unroll
      for (int r = 0; r < 8; ++r) { o[0][f][r] *= alpha[0][r]; o[1][f][r] *= alpha[1][r]; }
      o[0][f] = wmma_f16(pa0, bv, o[0][f]);
      o[1][f] = wmma_f16(pa1, bv, o[1][f]);
    }
  }

  // ---- epilogue: O /= l, store f32 ----
  float* Op = O + ((size_t)(b * LL + q0)) * DD + h * DKK;
#pragma unroll
  for (int rt = 0; rt < 2; ++rt)
#pragma unroll
    for (int r = 0; r < 8; ++r) {
      const float rinv = 1.0f / lacc[rt][r];
#pragma unroll
      for (int f = 0; f < 4; ++f) {
        Op[(size_t)(rt * 16 + rowhi + r) * DD + f * 16 + col] = o[rt][f][r] * rinv;
      }
    }
}

// ---------------------------------------------------------------------------
// Launcher
// ---------------------------------------------------------------------------
extern "C" void kernel_launch(void* const* d_in, const int* in_sizes, int n_in,
                              void* d_out, int out_size, void* d_ws, size_t ws_size,
                              hipStream_t stream) {
  const float* q    = (const float*)d_in[0];
  const float* k    = (const float*)d_in[1];
  const float* v    = (const float*)d_in[2];
  const float* wq_w = (const float*)d_in[3];
  const float* wq_b = (const float*)d_in[4];
  const float* wk_w = (const float*)d_in[5];
  const float* wk_b = (const float*)d_in[6];
  const float* wv_w = (const float*)d_in[7];
  const float* wv_b = (const float*)d_in[8];
  const float* wo_w = (const float*)d_in[9];
  const float* wo_b = (const float*)d_in[10];
  float* out = (float*)d_out;

  char* ws = (char*)d_ws;
  const size_t MB = (size_t)1 << 20;
  _Float16* Qh   = (_Float16*)(ws);             // 16 MB, f16 [B*L, D]
  _Float16* Kh   = (_Float16*)(ws + 16 * MB);   // 16 MB, f16 [B*L, D]
  _Float16* Vt   = (_Float16*)(ws + 32 * MB);   // 16 MB, f16 [B, D, L] (transposed)
  float*    Oa   = (float*)   (ws + 48 * MB);   // 32 MB, f32 [B*L, D]
  _Float16* wq16 = (_Float16*)(ws + 80 * MB);   // 2 MB each
  _Float16* wk16 = (_Float16*)(ws + 82 * MB);
  _Float16* wv16 = (_Float16*)(ws + 84 * MB);
  _Float16* wo16 = (_Float16*)(ws + 86 * MB);

  // Weight pre-conversion: removes fp32->f16 VALU from the GEMM hot loop
  const int wN = DD * DD;                       // 1,048,576
  const dim3 cGrid(wN / (256 * 8));
  cvt_f32_f16_kernel<<<cGrid, 256, 0, stream>>>(wq_w, wq16, wN);
  cvt_f32_f16_kernel<<<cGrid, 256, 0, stream>>>(wk_w, wk16, wN);
  cvt_f32_f16_kernel<<<cGrid, 256, 0, stream>>>(wv_w, wv16, wN);
  cvt_f32_f16_kernel<<<cGrid, 256, 0, stream>>>(wo_w, wo16, wN);

  const dim3 gBlk(256);
  const dim3 gGrid(MM / 256, DD / 128);         // (32, 8)

  gemm_wmma_kernel<0><<<gGrid, gBlk, 0, stream>>>(q, wq16, wq_b, Qh, nullptr, MM, DD, DD);
  gemm_wmma_kernel<0><<<gGrid, gBlk, 0, stream>>>(k, wk16, wk_b, Kh, nullptr, MM, DD, DD);
  gemm_wmma_kernel<1><<<gGrid, gBlk, 0, stream>>>(v, wv16, wv_b, Vt, nullptr, MM, DD, DD);

  const dim3 fGrid(LL / 256, BB * HH);          // (8, 64), 8 waves x 32 rows per block
  flash_attn_kernel<<<fGrid, gBlk, 0, stream>>>(Qh, Kh, Vt, Oa);

  gemm_wmma_kernel<2><<<gGrid, gBlk, 0, stream>>>(Oa, wo16, wo_b, nullptr, out, MM, DD, DD);
}
